// TGN_3667902071297
// MI455X (gfx1250) — compile-verified, matrix operand order checked by reading, not run on hardware
//
#include <hip/hip_runtime.h>
#include <cstdint>
#include <cstddef>

// ---------------------------------------------------------------------------
// CDNA5 / gfx1250 TGN forward: bf16 WMMA for all matmuls, f32 accumulate.
// Round 2: k_edge processes 32 edges/block (2 M-tiles) so each B fragment
// (weight tile from L2) is reused across 2 WMMAs -> 2x arithmetic intensity.
// ---------------------------------------------------------------------------

typedef __attribute__((ext_vector_type(16))) __bf16 v16bf;
typedef __attribute__((ext_vector_type(8)))  float  v8f;

__device__ __forceinline__ v8f wmma_bf16(v16bf a, v16bf b, v8f c) {
  // (neg_a, A, neg_b, B, c_mod, C, reuse_a, reuse_b)
  return __builtin_amdgcn_wmma_f32_16x16x32_bf16(false, a, false, b, (short)0, c, false, false);
}

#define KPAD1 576   // 548 padded to multiple of 32
#define EPB   32    // edges per block (two 16-row M tiles)

// ---------------- weight conversion: f32 row-major -> bf16 [rows][kpad] -----
__global__ void k_convert(const float* __restrict__ src, __bf16* __restrict__ dst,
                          int rows, int ksrc, int kpad) {
  int total = rows * kpad;
  for (int i = blockIdx.x * blockDim.x + threadIdx.x; i < total;
       i += gridDim.x * blockDim.x) {
    int r = i / kpad, k = i - r * kpad;
    float v = (k < ksrc) ? src[(size_t)r * ksrc + k] : 0.0f;
    dst[i] = (__bf16)v;
  }
}

// ---------------- zero scratch ---------------------------------------------
__global__ void k_zero(float* __restrict__ p, size_t n) {
  for (size_t i = blockIdx.x * (size_t)blockDim.x + threadIdx.x; i < n;
       i += (size_t)gridDim.x * blockDim.x) p[i] = 0.0f;
}

// ---------------- edge message MLP + scatter-add ---------------------------
// block = 256 threads (8 waves), one 32-edge tile per block.
__global__ __launch_bounds__(256) void k_edge(
    const float* __restrict__ nf, const float* __restrict__ ef,
    const float* __restrict__ et, const float* __restrict__ mem,
    const float* __restrict__ tww, const float* __restrict__ twb,
    const float* __restrict__ tbw, const float* __restrict__ tbb,
    const __bf16* __restrict__ W1b, const float* __restrict__ b1,
    const __bf16* __restrict__ W2b, const float* __restrict__ b2,
    const int* __restrict__ ei, float* __restrict__ agg,
    float* __restrict__ counts, int E_) {
  __shared__ __align__(32) __bf16 Amsg[EPB][KPAD1];   // 36,864 B
  __shared__ __align__(32) __bf16 hlds[EPB][256];     // 16,384 B
  __shared__ int s_src[EPB], s_dst[EPB];

  const int t     = threadIdx.x;
  const int ebase = blockIdx.x * EPB;

  if (t < EPB) {
    int e = ebase + t;
    s_src[t] = ei[e];
    s_dst[t] = ei[E_ + e];
    atomicAdd(&counts[s_dst[t]], 1.0f);
  }
  __syncthreads();

  // ---- build A (32 x 576 bf16): [nf_src|nf_dst|mem_src|mem_dst|ef|time|pad]
  {
    int e0 = t >> 4;          // 0..15
    int f0 = t & 15;          // feature phase (16 consecutive lanes -> coalesced)
    for (int half = 0; half < 2; ++half) {
      int e = e0 + half * 16;
      int src = s_src[e], dst = s_dst[e];
      float tv = et[ebase + e];
      for (int i = 0; i < KPAD1 / 16; ++i) {
        int f = f0 + i * 16;
        float v;
        if (f < 128)      v = nf[(size_t)src * 128 + f];
        else if (f < 256) v = nf[(size_t)dst * 128 + (f - 128)];
        else if (f < 384) v = mem[(size_t)src * 128 + (f - 256)];
        else if (f < 512) v = mem[(size_t)dst * 128 + (f - 384)];
        else if (f < 516) v = ef[(size_t)(ebase + e) * 4 + (f - 512)];
        else if (f < 548) {
          int j = f - 516;
          v = cosf(tv * (tww[j] + tbw[j]) + twb[j] + tbb[j]);
        } else v = 0.0f;
        Amsg[e][f] = (__bf16)v;
      }
    }
  }
  __syncthreads();

  const int lane  = t & 31;
  const int wave  = t >> 5;
  const int am    = lane & 15;          // A row / B col within tile
  const int akoff = (lane >> 4) * 16;   // K sub-block per half-wave
  const int nn    = lane & 15;
  const int mhi   = (lane >> 4) * 8;

  // ---- layer 1: h = relu(msg_in @ W1^T + b1)
  // N=256 -> 2 n-tiles/wave; M=32 -> 2 m-tiles. 4 WMMAs per k-step,
  // each B fragment reused across both m-tiles.
  {
    v8f acc00 = {}, acc01 = {}, acc10 = {}, acc11 = {};   // [mt][ntA/B]
    int n0a = (2 * wave) * 16;
    int n0b = n0a + 16;
    const __bf16* B0row = W1b + (size_t)(n0a + am) * KPAD1 + akoff;
    const __bf16* B1row = W1b + (size_t)(n0b + am) * KPAD1 + akoff;
    for (int kt = 0; kt < KPAD1 / 32; ++kt) {
      __builtin_prefetch(B0row + (kt + 1) * 32, 0, 1);
      v16bf a0 = *(const v16bf*)&Amsg[am][kt * 32 + akoff];        // m-tile 0
      v16bf a1 = *(const v16bf*)&Amsg[16 + am][kt * 32 + akoff];   // m-tile 1
      v16bf bA = *(const v16bf*)(B0row + kt * 32);
      v16bf bB = *(const v16bf*)(B1row + kt * 32);
      acc00 = wmma_bf16(a0, bA, acc00);
      acc10 = wmma_bf16(a1, bA, acc10);
      acc01 = wmma_bf16(a0, bB, acc01);
      acc11 = wmma_bf16(a1, bB, acc11);
    }
    float biasA = b1[n0a + nn], biasB = b1[n0b + nn];
    for (int r = 0; r < 8; ++r) {
      int m = r + mhi;
      hlds[m][n0a + nn]      = (__bf16)fmaxf(acc00[r] + biasA, 0.0f);
      hlds[m][n0b + nn]      = (__bf16)fmaxf(acc01[r] + biasB, 0.0f);
      hlds[16 + m][n0a + nn] = (__bf16)fmaxf(acc10[r] + biasA, 0.0f);
      hlds[16 + m][n0b + nn] = (__bf16)fmaxf(acc11[r] + biasB, 0.0f);
    }
  }
  __syncthreads();

  // ---- layer 2: msg = relu(h @ W2^T + b2), N=128 -> 1 n-tile/wave,
  // 2 m-tiles; B fragment reused across both. Then scatter-add into agg.
  {
    v8f acc0 = {}, acc1 = {};
    int n0 = wave * 16;
    const __bf16* Brow = W2b + (size_t)(n0 + am) * 256 + akoff;
    for (int kt = 0; kt < 256 / 32; ++kt) {
      v16bf a0 = *(const v16bf*)&hlds[am][kt * 32 + akoff];
      v16bf a1 = *(const v16bf*)&hlds[16 + am][kt * 32 + akoff];
      v16bf b  = *(const v16bf*)(Brow + kt * 32);
      acc0 = wmma_bf16(a0, b, acc0);
      acc1 = wmma_bf16(a1, b, acc1);
    }
    float bias = b2[n0 + nn];
    for (int r = 0; r < 8; ++r) {
      int m = r + mhi;
      float v0 = fmaxf(acc0[r] + bias, 0.0f);
      float v1 = fmaxf(acc1[r] + bias, 0.0f);
      atomicAdd(&agg[(size_t)s_dst[m] * 128 + (n0 + nn)], v0);       // L2 f32 atomic
      atomicAdd(&agg[(size_t)s_dst[16 + m] * 128 + (n0 + nn)], v1);
    }
  }
}

// ---------------- node update: GRU + embedding + classifier ----------------
// block = 256 threads (8 waves), one 16-node tile per block. 56 KB LDS arena.
__global__ __launch_bounds__(256) void k_node(
    const float* __restrict__ nf, const float* __restrict__ mem,
    const float* __restrict__ agg, const float* __restrict__ counts,
    const __bf16* __restrict__ Wihb, const __bf16* __restrict__ Whhb,
    const float* __restrict__ bih, const float* __restrict__ bhh,
    const __bf16* __restrict__ We1b, const float* __restrict__ be1,
    const __bf16* __restrict__ We2b, const float* __restrict__ be2,
    const __bf16* __restrict__ Wc1b, const float* __restrict__ bc1,
    const float* __restrict__ Wc2, const float* __restrict__ bc2,
    float* __restrict__ out) {
  __shared__ __align__(32) unsigned char smem[57344];
  // phase-1 views
  __bf16 (*Aagg)[128] = (__bf16(*)[128])(smem);            // 4 KB
  __bf16 (*Amem)[128] = (__bf16(*)[128])(smem + 4096);     // 4 KB
  float  (*gi)[384]   = (float (*)[384])(smem + 8192);     // 24 KB
  float  (*gh)[384]   = (float (*)[384])(smem + 32768);    // 24 KB
  // phase-2 overlays (previous views dead after barriers)
  __bf16 (*A0)[256]   = (__bf16(*)[256])(smem);            // over Aagg/Amem
  __bf16 (*e1)[128]   = (__bf16(*)[128])(smem + 8192);     // over gi
  __bf16 (*e2)[128]   = (__bf16(*)[128])(smem + 12288);
  float  (*cl)[64]    = (float (*)[64]) (smem + 16384);

  const int t     = threadIdx.x;
  const int nodeb = blockIdx.x * 16;
  const int lane  = t & 31;
  const int wave  = t >> 5;
  const int am    = lane & 15;
  const int akoff = (lane >> 4) * 16;
  const int nn    = lane & 15;
  const int mhi   = (lane >> 4) * 8;

  // ---- build A tiles: segment-mean(agg) and memory, bf16
  for (int idx = t; idx < 16 * 128; idx += 256) {
    int m = idx >> 7, k = idx & 127;
    int node = nodeb + m;
    float cnt = fmaxf(counts[node], 1.0f);
    Aagg[m][k] = (__bf16)(agg[(size_t)node * 128 + k] / cnt);
    Amem[m][k] = (__bf16)(mem[(size_t)node * 128 + k]);
  }
  __syncthreads();

  // ---- GRU matmuls: gi = mean @ Wih^T + bih ; gh = memory @ Whh^T + bhh
  for (int job = wave; job < 48; job += 8) {       // 24 gi tiles + 24 gh tiles
    bool isGi = job < 24;
    int nt = isGi ? job : job - 24;
    int n0 = nt * 16;
    const __bf16 (*A)[128] = isGi ? Aagg : Amem;
    const __bf16* W = isGi ? Wihb : Whhb;
    v8f acc = {};
    const __bf16* Brow = W + (size_t)(n0 + am) * 128 + akoff;
    for (int kt = 0; kt < 4; ++kt) {
      v16bf a = *(const v16bf*)&A[am][kt * 32 + akoff];
      v16bf b = *(const v16bf*)(Brow + kt * 32);
      acc = wmma_bf16(a, b, acc);
    }
    const float* bias = isGi ? bih : bhh;
    float bv = bias[n0 + nn];
    float (*G)[384] = isGi ? gi : gh;
    for (int r = 0; r < 8; ++r) G[r + mhi][n0 + nn] = acc[r] + bv;
  }
  __syncthreads();

  // ---- gates elementwise; build A0 = [nf | new_memory] bf16
  for (int idx = t; idx < 16 * 128; idx += 256) {
    int m = idx >> 7, k = idx & 127;
    int node = nodeb + m;
    float ir = gi[m][k],       hr = gh[m][k];
    float iz = gi[m][128 + k], hz = gh[m][128 + k];
    float in_ = gi[m][256 + k], hn = gh[m][256 + k];
    float r = 1.0f / (1.0f + __expf(-(ir + hr)));
    float z = 1.0f / (1.0f + __expf(-(iz + hz)));
    float n = tanhf(in_ + r * hn);
    float mv = mem[(size_t)node * 128 + k];
    float nm = (1.0f - z) * n + z * mv;
    A0[m][k]       = (__bf16)nf[(size_t)node * 128 + k];
    A0[m][128 + k] = (__bf16)nm;
  }
  __syncthreads();

  // ---- e1 = relu(A0 @ We1^T + be1), N=128, K=256
  {
    v8f acc = {};
    int n0 = wave * 16;
    const __bf16* Brow = We1b + (size_t)(n0 + am) * 256 + akoff;
    for (int kt = 0; kt < 8; ++kt) {
      v16bf a = *(const v16bf*)&A0[am][kt * 32 + akoff];
      v16bf b = *(const v16bf*)(Brow + kt * 32);
      acc = wmma_bf16(a, b, acc);
    }
    float bv = be1[n0 + nn];
    for (int r = 0; r < 8; ++r)
      e1[r + mhi][n0 + nn] = (__bf16)fmaxf(acc[r] + bv, 0.0f);
  }
  __syncthreads();

  // ---- e2 = relu(e1 @ We2^T + be2), N=128, K=128
  {
    v8f acc = {};
    int n0 = wave * 16;
    const __bf16* Brow = We2b + (size_t)(n0 + am) * 128 + akoff;
    for (int kt = 0; kt < 4; ++kt) {
      v16bf a = *(const v16bf*)&e1[am][kt * 32 + akoff];
      v16bf b = *(const v16bf*)(Brow + kt * 32);
      acc = wmma_bf16(a, b, acc);
    }
    float bv = be2[n0 + nn];
    for (int r = 0; r < 8; ++r)
      e2[r + mhi][n0 + nn] = (__bf16)fmaxf(acc[r] + bv, 0.0f);
  }
  __syncthreads();

  // ---- c = relu(e2 @ Wc1^T + bc1), N=64 (waves 0..3, each wave fully active)
  if (wave < 4) {
    v8f acc = {};
    int n0 = wave * 16;
    const __bf16* Brow = Wc1b + (size_t)(n0 + am) * 128 + akoff;
    for (int kt = 0; kt < 4; ++kt) {
      v16bf a = *(const v16bf*)&e2[am][kt * 32 + akoff];
      v16bf b = *(const v16bf*)(Brow + kt * 32);
      acc = wmma_bf16(a, b, acc);
    }
    float bv = bc1[n0 + nn];
    for (int r = 0; r < 8; ++r)
      cl[r + mhi][n0 + nn] = fmaxf(acc[r] + bv, 0.0f);
  }
  __syncthreads();

  // ---- logits = c @ Wc2^T + bc2  (16 nodes x 2 classes, scalar f32)
  if (t < 32) {
    int m = t >> 1, cls = t & 1;
    float s = bc2[cls];
    for (int k = 0; k < 64; ++k) s += cl[m][k] * Wc2[cls * 64 + k];
    out[(size_t)(nodeb + m) * 2 + cls] = s;
  }
}

// ---------------------------------------------------------------------------
extern "C" void kernel_launch(void* const* d_in, const int* in_sizes, int n_in,
                              void* d_out, int out_size, void* d_ws, size_t ws_size,
                              hipStream_t stream) {
  const float* nf  = (const float*)d_in[0];
  const float* ef  = (const float*)d_in[1];
  const float* et  = (const float*)d_in[2];
  const float* mem = (const float*)d_in[3];
  const float* tww = (const float*)d_in[4];
  const float* twb = (const float*)d_in[5];
  const float* tbw = (const float*)d_in[6];
  const float* tbb = (const float*)d_in[7];
  const float* W1  = (const float*)d_in[8];
  const float* b1  = (const float*)d_in[9];
  const float* W2  = (const float*)d_in[10];
  const float* b2  = (const float*)d_in[11];
  const float* Wih = (const float*)d_in[12];
  const float* Whh = (const float*)d_in[13];
  const float* bih = (const float*)d_in[14];
  const float* bhh = (const float*)d_in[15];
  const float* We1 = (const float*)d_in[16];
  const float* be1 = (const float*)d_in[17];
  const float* We2 = (const float*)d_in[18];
  const float* be2 = (const float*)d_in[19];
  const float* Wc1 = (const float*)d_in[20];
  const float* bc1 = (const float*)d_in[21];
  const float* Wc2 = (const float*)d_in[22];
  const float* bc2 = (const float*)d_in[23];
  const int*   ei  = (const int*)d_in[24];

  const int NN = in_sizes[0] / 128;   // nodes (100000)
  const int EE = in_sizes[2];         // edges (500000)

  // workspace layout: bf16 weights, then counts + agg (contiguous, zeroed)
  char* ws = (char*)d_ws;
  __bf16* W1b  = (__bf16*)ws; ws += (size_t)256 * KPAD1 * 2;
  __bf16* W2b  = (__bf16*)ws; ws += (size_t)128 * 256 * 2;
  __bf16* Wihb = (__bf16*)ws; ws += (size_t)384 * 128 * 2;
  __bf16* Whhb = (__bf16*)ws; ws += (size_t)384 * 128 * 2;
  __bf16* We1b = (__bf16*)ws; ws += (size_t)128 * 256 * 2;
  __bf16* We2b = (__bf16*)ws; ws += (size_t)128 * 128 * 2;
  __bf16* Wc1b = (__bf16*)ws; ws += (size_t)64 * 128 * 2;
  ws = (char*)(((uintptr_t)ws + 255) & ~(uintptr_t)255);
  float* counts = (float*)ws; ws += (size_t)NN * 4;
  float* agg    = (float*)ws;                       // NN*128 floats

  // 1) bf16 weight conversion (idempotent, re-run every call)
  k_convert<<<128, 256, 0, stream>>>(W1,  W1b,  256, 548, KPAD1);
  k_convert<<<64,  256, 0, stream>>>(W2,  W2b,  128, 256, 256);
  k_convert<<<64,  256, 0, stream>>>(Wih, Wihb, 384, 128, 128);
  k_convert<<<64,  256, 0, stream>>>(Whh, Whhb, 384, 128, 128);
  k_convert<<<64,  256, 0, stream>>>(We1, We1b, 128, 256, 256);
  k_convert<<<32,  256, 0, stream>>>(We2, We2b, 128, 128, 128);
  k_convert<<<16,  256, 0, stream>>>(Wc1, Wc1b, 64,  128, 128);

  // 2) zero counts + agg (contiguous NN*129 floats)
  k_zero<<<2048, 256, 0, stream>>>(counts, (size_t)NN * 129);

  // 3) edge message MLP + segment scatter-add (one 32-edge tile / block)
  k_edge<<<EE / EPB, 256, 0, stream>>>(nf, ef, et, mem, tww, twb, tbw, tbb,
                                       W1b, b1, W2b, b2, ei, agg, counts, EE);

  // 4) fused GRU + embedding + classifier (one 16-node tile / block)
  k_node<<<NN / 16, 256, 0, stream>>>(nf, mem, agg, counts,
                                      Wihb, Whhb, bih, bhh,
                                      We1b, be1, We2b, be2,
                                      Wc1b, bc1, Wc2, bc2,
                                      (float*)d_out);
}
